// GATEdgeConvHybrid_52390011076911
// MI455X (gfx1250) — matrix operand order repaired and checked by reading, not compiled, not run on hardware
//
#include <hip/hip_runtime.h>

typedef _Float16 half_t;
typedef __attribute__((ext_vector_type(16))) _Float16 v16h;
typedef __attribute__((ext_vector_type(8)))  _Float16 v8h;
typedef __attribute__((ext_vector_type(8)))  float    v8f;
typedef __attribute__((ext_vector_type(4)))  float    v4f;

#define GHEADS 4

static __device__ __forceinline__ v8f wmma_f16(v16h a, v16h b, v8f c) {
  // D = A(16x32 f16) * B(32x16 f16) + C(16x16 f32)
  return __builtin_amdgcn_wmma_f32_16x16x32_f16(false, a, false, b, (short)0, c, false, false);
}

// order-preserving float <-> uint key for atomicMax on floats
static __device__ __forceinline__ unsigned fkey(float f) {
  unsigned u = __float_as_uint(f);
  return u ^ ((u & 0x80000000u) ? 0xFFFFFFFFu : 0x80000000u);
}
static __device__ __forceinline__ float funkey(unsigned k) {
  unsigned u = (k & 0x80000000u) ? (k ^ 0x80000000u) : ~k;
  return __uint_as_float(u);
}

// B fragment fetch from pre-swizzled arena: one contiguous 32B load per lane.
static __device__ __forceinline__ v16h load_bfrag(const half_t* __restrict__ Wp,
                                                  int kb, int nt, int ntiles, int lane) {
  return *(const v16h*)(Wp + (((size_t)kb * ntiles + nt) * 32 + lane) * 16);
}

// ---------------------------------------------------------------------------
// Pack f32 weight matrix [K][NOUT] into WMMA-B fragment order (f16):
// Wp[((kb*ntiles + nt)*32 + lane)*16 + j] = W[kb*32 + (lane>>4)*8 + (j&7) + (j>>3)*16][nt*16 + (lane&15)]
// ---------------------------------------------------------------------------
__global__ void pack_b_kernel(const float* __restrict__ W, half_t* __restrict__ Wp,
                              int K, int NOUT) {
  int i = blockIdx.x * blockDim.x + threadIdx.x;
  if (i >= K * NOUT) return;
  int j    = i & 15;
  int lane = (i >> 4) & 31;
  int rest = i >> 9;
  int ntiles = NOUT >> 4;
  int nt = rest % ntiles;
  int kb = rest / ntiles;
  int row = kb * 32 + ((lane >> 4) << 3) + (j & 7) + ((j >> 3) << 4);
  int col = nt * 16 + (lane & 15);
  Wp[i] = (half_t)W[(size_t)row * NOUT + col];
}

// ---------------------------------------------------------------------------
// EdgeConv pass 1: h1 = [x_i, x_j - x_i] @ W1 + b1  (gather + WMMA GEMM)
// + per-column BN sum/sumsq via LDS. One wave = 16 edges x 64 cols, K=2*DIN.
// ---------------------------------------------------------------------------
template<int DIN>
__global__ void ec_lin1_kernel(const float* __restrict__ X,
                               const int* __restrict__ src,
                               const int* __restrict__ dst,
                               const half_t* __restrict__ Wp,   // packed [2*DIN][64]
                               const float* __restrict__ bias,  // [64]
                               half_t* __restrict__ Hout,       // [E][64]
                               float* __restrict__ gsum,
                               float* __restrict__ gsq,
                               int E, int tpw)
{
  constexpr int KB = (2 * DIN) / 32;
  __shared__ float s_sum[64];
  __shared__ float s_sq[64];
  const int tid = threadIdx.x;
  if (tid < 64) { s_sum[tid] = 0.f; s_sq[tid] = 0.f; }
  __syncthreads();

  const int lane = tid & 31;
  const int col  = lane & 15;
  const int hi   = lane >> 4;
  const int akb  = hi << 3;   // A K-base within 32-block: 0 or 8

  // preload B fragments (vector loads from packed arena)
  v16h bfrag[KB][4];
  #pragma unroll
  for (int kb = 0; kb < KB; ++kb)
    #pragma unroll
    for (int nt = 0; nt < 4; ++nt)
      bfrag[kb][nt] = load_bfrag(Wp, kb, nt, 4, lane);

  // hoisted bias
  float bv[4];
  #pragma unroll
  for (int nt = 0; nt < 4; ++nt) bv[nt] = bias[nt * 16 + col];

  const int totTiles = E >> 4;
  const int gw = blockIdx.x * (blockDim.x >> 5) + (tid >> 5);
  const int t0 = gw * tpw;
  for (int t = t0; t < t0 + tpw; ++t) {
    if (t >= totTiles) break;            // wave-uniform
    const int e0 = t << 4;
    const int e  = e0 + col;
    const float* xi = X + (size_t)dst[e] * DIN;
    const float* xj = X + (size_t)src[e] * DIN;

    v8f acc[4] = {};
    #pragma unroll
    for (int kb = 0; kb < KB; ++kb) {
      v16h a;
      #pragma unroll
      for (int run = 0; run < 2; ++run) {
        const int kk0 = kb * 32 + akb + run * 16;   // 8-elem run, 32B aligned
        v4f f0, f1;
        if (kb * 32 + run * 16 < DIN) {             // uniform: run fully in x_i part
          f0 = *(const v4f*)(xi + kk0);
          f1 = *(const v4f*)(xi + kk0 + 4);
        } else {                                    // run fully in (x_j - x_i) part
          const int k2 = kk0 - DIN;
          v4f i0 = *(const v4f*)(xi + k2), i1 = *(const v4f*)(xi + k2 + 4);
          v4f j0 = *(const v4f*)(xj + k2), j1 = *(const v4f*)(xj + k2 + 4);
          f0 = j0 - i0; f1 = j1 - i1;
        }
        #pragma unroll
        for (int q = 0; q < 4; ++q) {
          a[run * 8 + q]     = (half_t)f0[q];
          a[run * 8 + 4 + q] = (half_t)f1[q];
        }
      }
      #pragma unroll
      for (int nt = 0; nt < 4; ++nt)
        acc[nt] = wmma_f16(a, bfrag[kb][nt], acc[nt]);
    }

    #pragma unroll
    for (int nt = 0; nt < 4; ++nt) {
      const int c = nt * 16 + col;
      half_t* op = Hout + (size_t)(e0 + (hi << 3)) * 64 + c;
      float ps = 0.f, pq = 0.f;
      #pragma unroll
      for (int r = 0; r < 8; ++r) {
        const float v = acc[nt][r] + bv[nt];
        op[r * 64] = (half_t)v;
        ps += v; pq += v * v;
      }
      atomicAdd(&s_sum[c], ps);
      atomicAdd(&s_sq[c], pq);
    }
  }
  __syncthreads();
  if (tid < 64) { atomicAdd(&gsum[tid], s_sum[tid]); atomicAdd(&gsq[tid], s_sq[tid]); }
}

// ---------------------------------------------------------------------------
// EdgeConv passes 2/3: A = relu(Hin*scale + shift) (BN folded), GEMM with W.
// SCATTER=false: f16 Hout + BN stats.  SCATTER=true: atomic segment-sum
// of (acc + bias) into agg[dst].
// ---------------------------------------------------------------------------
template<bool SCATTER>
__global__ void ec_lin23_kernel(const half_t* __restrict__ Hin, // [E][64]
                                const float* __restrict__ scale,
                                const float* __restrict__ shift,
                                const half_t* __restrict__ Wp,  // packed [64][64]
                                const float* __restrict__ bias,
                                half_t* __restrict__ Hout,
                                float* __restrict__ gsum,
                                float* __restrict__ gsq,
                                const int* __restrict__ dst,
                                float* __restrict__ agg,        // [N][64]
                                int E, int tpw)
{
  constexpr int KB = 2;
  __shared__ float s_sum[64];
  __shared__ float s_sq[64];
  const int tid = threadIdx.x;
  if constexpr (!SCATTER) {
    if (tid < 64) { s_sum[tid] = 0.f; s_sq[tid] = 0.f; }
    __syncthreads();
  }

  const int lane = tid & 31;
  const int col  = lane & 15;
  const int hi   = lane >> 4;
  const int akb  = hi << 3;

  v16h bfrag[KB][4];
  #pragma unroll
  for (int kb = 0; kb < KB; ++kb)
    #pragma unroll
    for (int nt = 0; nt < 4; ++nt)
      bfrag[kb][nt] = load_bfrag(Wp, kb, nt, 4, lane);

  // hoisted BN affine (registers, vector loads) and bias
  float sc[KB][16], sh[KB][16];
  #pragma unroll
  for (int kb = 0; kb < KB; ++kb)
    #pragma unroll
    for (int run = 0; run < 2; ++run) {
      const int kk0 = kb * 32 + akb + run * 16;
      v4f s0 = *(const v4f*)(scale + kk0), s1 = *(const v4f*)(scale + kk0 + 4);
      v4f h0 = *(const v4f*)(shift + kk0), h1 = *(const v4f*)(shift + kk0 + 4);
      #pragma unroll
      for (int q = 0; q < 4; ++q) {
        sc[kb][run * 8 + q] = s0[q]; sc[kb][run * 8 + 4 + q] = s1[q];
        sh[kb][run * 8 + q] = h0[q]; sh[kb][run * 8 + 4 + q] = h1[q];
      }
    }
  float bv[4];
  #pragma unroll
  for (int nt = 0; nt < 4; ++nt) bv[nt] = bias[nt * 16 + col];

  const int totTiles = E >> 4;
  const int gw = blockIdx.x * (blockDim.x >> 5) + (tid >> 5);
  const int t0 = gw * tpw;
  for (int t = t0; t < t0 + tpw; ++t) {
    if (t >= totTiles) break;
    const int e0 = t << 4;
    const half_t* hrow = Hin + (size_t)(e0 + col) * 64;
    if (t + 1 < totTiles)
      __builtin_prefetch(hrow + 16 * 64, 0, 0);   // next tile's row for this lane

    v8f acc[4] = {};
    #pragma unroll
    for (int kb = 0; kb < KB; ++kb) {
      const v8h h0 = *(const v8h*)(hrow + kb * 32 + akb);
      const v8h h1 = *(const v8h*)(hrow + kb * 32 + akb + 16);
      v16h a;
      #pragma unroll
      for (int j = 0; j < 8; ++j) {
        a[j]     = (half_t)fmaxf(fmaf((float)h0[j], sc[kb][j],     sh[kb][j]),     0.f);
        a[8 + j] = (half_t)fmaxf(fmaf((float)h1[j], sc[kb][8 + j], sh[kb][8 + j]), 0.f);
      }
      #pragma unroll
      for (int nt = 0; nt < 4; ++nt)
        acc[nt] = wmma_f16(a, bfrag[kb][nt], acc[nt]);
    }

    if constexpr (SCATTER) {
      int drows[8];
      #pragma unroll
      for (int r = 0; r < 8; ++r) drows[r] = dst[e0 + (hi << 3) + r];
      #pragma unroll
      for (int nt = 0; nt < 4; ++nt) {
        const int c = nt * 16 + col;
        #pragma unroll
        for (int r = 0; r < 8; ++r)
          atomicAdd(&agg[(size_t)drows[r] * 64 + c], acc[nt][r] + bv[nt]);
      }
    } else {
      #pragma unroll
      for (int nt = 0; nt < 4; ++nt) {
        const int c = nt * 16 + col;
        half_t* op = Hout + (size_t)(e0 + (hi << 3)) * 64 + c;
        float ps = 0.f, pq = 0.f;
        #pragma unroll
        for (int r = 0; r < 8; ++r) {
          const float v = acc[nt][r] + bv[nt];
          op[r * 64] = (half_t)v;
          ps += v; pq += v * v;
        }
        atomicAdd(&s_sum[c], ps);
        atomicAdd(&s_sq[c], pq);
      }
    }
  }
  if constexpr (!SCATTER) {
    __syncthreads();
    if (tid < 64) { atomicAdd(&gsum[tid], s_sum[tid]); atomicAdd(&gsq[tid], s_sq[tid]); }
  }
}

// ---------------------------------------------------------------------------
// Generic node GEMM: out[N][NOUT] = X[N][DIN] @ W[DIN][NOUT] (+ bias)
// NOUT is a template parameter so row strides fold into store offsets.
// ---------------------------------------------------------------------------
template<int DIN, int NOUT, bool HB>
__global__ void node_gemm_kernel(const float* __restrict__ X,
                                 const half_t* __restrict__ Wp,  // packed
                                 const float* __restrict__ bias,
                                 float* __restrict__ out,
                                 int N)
{
  constexpr int KB = DIN / 32;
  constexpr int NT = NOUT / 16;
  const int lane = threadIdx.x & 31;
  const int col  = lane & 15;
  const int hi   = lane >> 4;
  const int akb  = hi << 3;
  const int t = blockIdx.x * (blockDim.x >> 5) + (threadIdx.x >> 5);
  if (t >= (N >> 4)) return;

  const float* xp = X + (size_t)((t << 4) + col) * DIN;
  v16h afrag[KB];
  #pragma unroll
  for (int kb = 0; kb < KB; ++kb) {
    v16h a;
    #pragma unroll
    for (int run = 0; run < 2; ++run) {
      const int kk0 = kb * 32 + akb + run * 16;
      v4f f0 = *(const v4f*)(xp + kk0);
      v4f f1 = *(const v4f*)(xp + kk0 + 4);
      #pragma unroll
      for (int q = 0; q < 4; ++q) {
        a[run * 8 + q]     = (half_t)f0[q];
        a[run * 8 + 4 + q] = (half_t)f1[q];
      }
    }
    afrag[kb] = a;
  }

  float bv[NT];
  #pragma unroll
  for (int nt = 0; nt < NT; ++nt) bv[nt] = HB ? bias[nt * 16 + col] : 0.f;

  float* obase = out + (size_t)((t << 4) + (hi << 3)) * NOUT + col;
  #pragma unroll
  for (int nt = 0; nt < NT; ++nt) {
    v8f acc = {};
    #pragma unroll
    for (int kb = 0; kb < KB; ++kb)
      acc = wmma_f16(afrag[kb], load_bfrag(Wp, kb, nt, NT, lane), acc);
    float* op = obase + nt * 16;
    #pragma unroll
    for (int r = 0; r < 8; ++r)
      op[r * NOUT] = acc[r] + bv[nt];      // constant-offset stores
  }
}

// ---------------------------------------------------------------------------
// Small support kernels
// ---------------------------------------------------------------------------
__global__ void bn_finalize_kernel(const float* __restrict__ sum,
                                   const float* __restrict__ sq,
                                   const float* __restrict__ g,
                                   const float* __restrict__ be,
                                   float* __restrict__ scale,
                                   float* __restrict__ shift, float inv_cnt)
{
  int c = threadIdx.x;  // 64 threads
  float mean = sum[c] * inv_cnt;
  float var  = sq[c] * inv_cnt - mean * mean;
  float s = g[c] * rsqrtf(var + 1e-5f);
  scale[c] = s;
  shift[c] = be[c] - mean * s;
}

__global__ void deg_kernel(const int* __restrict__ dst, float* __restrict__ deg, int E) {
  int e = blockIdx.x * blockDim.x + threadIdx.x;
  if (e < E) atomicAdd(&deg[dst[e]], 1.f);
}

// relu(agg [* 1/deg]) [+ res] on [N][64], vectorized x4
__global__ void agg_fin_kernel(const float* __restrict__ agg,
                               const float* __restrict__ deg,   // null => sum aggr
                               const float* __restrict__ res,   // nullable residual
                               float* __restrict__ out, int N)
{
  int i4 = blockIdx.x * blockDim.x + threadIdx.x;
  if (i4 >= N * 16) return;
  int n = i4 >> 4;
  size_t off = (size_t)i4 * 4;
  v4f v = *(const v4f*)(agg + off);
  if (deg) {
    float inv = 1.f / fmaxf(deg[n], 1.f);
    v *= inv;
  }
  #pragma unroll
  for (int q = 0; q < 4; ++q) v[q] = fmaxf(v[q], 0.f);
  if (res) v += *(const v4f*)(res + off);
  *(v4f*)(out + off) = v;
}

// ---------------------------------------------------------------------------
// GAT kernels
// ---------------------------------------------------------------------------
__global__ void gat_att_kernel(const float* __restrict__ H,       // [N][256]
                               const float* __restrict__ att_src, // [4][64]
                               const float* __restrict__ att_dst,
                               float* __restrict__ asrc, float* __restrict__ adst,
                               float* __restrict__ eself, unsigned* __restrict__ emaxk,
                               int N)
{
  int i = blockIdx.x * blockDim.x + threadIdx.x;
  if (i >= N * GHEADS) return;
  int n = i >> 2, hd = i & 3;
  const float* hp = H + (size_t)n * 256 + hd * 64;
  const float* as = att_src + hd * 64;
  const float* ad = att_dst + hd * 64;
  float s = 0.f, d = 0.f;
  #pragma unroll 4
  for (int f = 0; f < 16; ++f) {
    v4f hv = *(const v4f*)(hp + f * 4);
    v4f av = *(const v4f*)(as + f * 4);
    v4f dv = *(const v4f*)(ad + f * 4);
    #pragma unroll
    for (int q = 0; q < 4; ++q) { s += hv[q] * av[q]; d += hv[q] * dv[q]; }
  }
  asrc[i] = s; adst[i] = d;
  float e = s + d; e = e > 0.f ? e : 0.2f * e;   // self-loop logit
  eself[i] = e;
  emaxk[i] = fkey(e);                            // init max with self-loop
}

__global__ void gat_edge_max_kernel(const int* __restrict__ src, const int* __restrict__ dst,
                                    const float* __restrict__ asrc, const float* __restrict__ adst,
                                    float* __restrict__ ew, unsigned* __restrict__ emaxk, int E)
{
  int i = blockIdx.x * blockDim.x + threadIdx.x;
  if (i >= E * GHEADS) return;
  int e = i >> 2, hd = i & 3;
  float v = asrc[src[e] * 4 + hd] + adst[dst[e] * 4 + hd];
  v = v > 0.f ? v : 0.2f * v;
  ew[i] = v;
  atomicMax(&emaxk[dst[e] * 4 + hd], fkey(v));
}

__global__ void gat_selfdenom_kernel(float* __restrict__ eself_pself,
                                     const unsigned* __restrict__ emaxk,
                                     float* __restrict__ denom, int N)
{
  int i = blockIdx.x * blockDim.x + threadIdx.x;
  if (i >= N * GHEADS) return;
  float p = expf(eself_pself[i] - funkey(emaxk[i]));
  eself_pself[i] = p;
  denom[i] = p;
}

__global__ void gat_edge_exp_kernel(float* __restrict__ ew, const unsigned* __restrict__ emaxk,
                                    const int* __restrict__ dst, float* __restrict__ denom, int E)
{
  int i = blockIdx.x * blockDim.x + threadIdx.x;
  if (i >= E * GHEADS) return;
  int e = i >> 2, hd = i & 3;
  int dh = dst[e] * 4 + hd;
  float p = expf(ew[i] - funkey(emaxk[dh]));
  ew[i] = p;
  atomicAdd(&denom[dh], p);
}

// out[n][:] = H[n][:] * alpha_self, vectorized x4: grid = N, block = 64
__global__ void gat_selfout_kernel(const float* __restrict__ H, const float* __restrict__ pself,
                                   const float* __restrict__ denom, float* __restrict__ out, int N)
{
  int n = blockIdx.x, c4 = threadIdx.x;  // c4 in [0,64)
  int c = c4 * 4;
  int hd = c >> 6;
  float a = pself[n * 4 + hd] / (denom[n * 4 + hd] + 1e-16f);
  size_t off = (size_t)n * 256 + c;
  *(v4f*)(out + off) = *(const v4f*)(H + off) * a;
}

__global__ void gat_edge_aggr_kernel(const float* __restrict__ H, const float* __restrict__ ew,
                                     const float* __restrict__ denom,
                                     const int* __restrict__ src, const int* __restrict__ dst,
                                     float* __restrict__ out, int E)
{
  int e = blockIdx.x, c = threadIdx.x;   // grid = E, block = 256
  int hd = c >> 6;
  int s = src[e], d = dst[e];
  float a = ew[e * 4 + hd] / (denom[d * 4 + hd] + 1e-16f);
  atomicAdd(&out[(size_t)d * 256 + c], H[(size_t)s * 256 + c] * a);
}

// elu(out + bias), concat heads; vectorized x4: grid = N, block = 64
__global__ void gat_final_concat_kernel(const float* __restrict__ out, const float* __restrict__ bias,
                                        float* __restrict__ g, int N)
{
  int n = blockIdx.x, c4 = threadIdx.x;
  int c = c4 * 4;
  v4f v = *(const v4f*)(out + (size_t)n * 256 + c) + *(const v4f*)(bias + c);
  #pragma unroll
  for (int q = 0; q < 4; ++q) v[q] = v[q] > 0.f ? v[q] : expf(v[q]) - 1.f;
  *(v4f*)(g + (size_t)n * 256 + c) = v;
}

// elu(mean-over-heads(out) + bias); vectorized x4 over [N][64]
__global__ void gat_final_mean_kernel(const float* __restrict__ out, const float* __restrict__ bias,
                                      float* __restrict__ g, int N)
{
  int i4 = blockIdx.x * blockDim.x + threadIdx.x;
  if (i4 >= N * 16) return;
  int n = i4 >> 4, f = (i4 & 15) * 4;
  const float* op = out + (size_t)n * 256;
  v4f v = (*(const v4f*)(op + f) + *(const v4f*)(op + 64 + f) +
           *(const v4f*)(op + 128 + f) + *(const v4f*)(op + 192 + f)) * 0.25f +
          *(const v4f*)(bias + f);
  #pragma unroll
  for (int q = 0; q < 4; ++q) v[q] = v[q] > 0.f ? v[q] : expf(v[q]) - 1.f;
  *(v4f*)(g + (size_t)n * 64 + f) = v;
}

// ---------------------------------------------------------------------------
// Host launch
// ---------------------------------------------------------------------------
extern "C" void kernel_launch(void* const* d_in, const int* in_sizes, int n_in,
                              void* d_out, int out_size, void* d_ws, size_t ws_size,
                              hipStream_t stream)
{
  const int N = in_sizes[0] / 32;
  const int E = in_sizes[1] / 2;

  const float* x   = (const float*)d_in[0];
  const int*   ei  = (const int*)d_in[1];
  const int*   src = ei;
  const int*   dst = ei + E;

  int ai = 2;
  const float* scW = (const float*)d_in[ai++];
  const float* scb = (const float*)d_in[ai++];
  struct MLP { const float *W1,*b1,*g1,*be1,*W2,*b2,*g2,*be2,*W3,*b3; };
  auto readMLP = [&]() {
    MLP m;
    m.W1=(const float*)d_in[ai++]; m.b1=(const float*)d_in[ai++];
    m.g1=(const float*)d_in[ai++]; m.be1=(const float*)d_in[ai++];
    m.W2=(const float*)d_in[ai++]; m.b2=(const float*)d_in[ai++];
    m.g2=(const float*)d_in[ai++]; m.be2=(const float*)d_in[ai++];
    m.W3=(const float*)d_in[ai++]; m.b3=(const float*)d_in[ai++];
    return m;
  };
  MLP ec1 = readMLP(), ec2 = readMLP(), ec3 = readMLP();
  struct GATP { const float *W,*as,*ad,*b; };
  GATP g1p { (const float*)d_in[ai], (const float*)d_in[ai+1], (const float*)d_in[ai+2], (const float*)d_in[ai+3] }; ai += 4;
  GATP g2p { (const float*)d_in[ai], (const float*)d_in[ai+1], (const float*)d_in[ai+2], (const float*)d_in[ai+3] }; ai += 4;
  const float* clfW = (const float*)d_in[ai++];
  const float* clfb = (const float*)d_in[ai++];

  // ---- workspace carving ----
  char* base = (char*)d_ws;
  size_t cur = 0;
  auto alloc = [&](size_t bytes) -> void* {
    void* p = base + cur;
    cur += (bytes + 255) & ~(size_t)255;
    return p;
  };

  half_t* hw = (half_t*)alloc(512 * 1024);
  size_t hoff = 0;
  auto halloc = [&](size_t n) { half_t* p = hw + hoff; hoff += n; return p; };
  half_t* scWh  = halloc(32 * 64);
  half_t* e1W1h = halloc(64 * 64),  *e1W2h = halloc(64 * 64), *e1W3h = halloc(64 * 64);
  half_t* e2W1h = halloc(128 * 64), *e2W2h = halloc(64 * 64), *e2W3h = halloc(64 * 64);
  half_t* e3W1h = halloc(128 * 64), *e3W2h = halloc(64 * 64), *e3W3h = halloc(64 * 64);
  half_t* g1Wh  = halloc(64 * 256);
  half_t* g2Wh  = halloc(256 * 256);
  half_t* clfWh = halloc(64 * 16);

  float* X_IN = (float*)alloc((size_t)N * 64 * 4);
  float* CUR1 = (float*)alloc((size_t)N * 64 * 4);
  float* CUR2 = (float*)alloc((size_t)N * 64 * 4);
  float* CUR3 = (float*)alloc((size_t)N * 64 * 4);
  float* AGG  = (float*)alloc((size_t)N * 64 * 4);
  float* DEG  = (float*)alloc((size_t)N * 4);
  float* STATS= (float*)alloc(512 * 4);  // sum1 sq1 scale1 shift1 sum2 sq2 scale2 shift2
  float* SUM1 = STATS, *SQ1 = STATS+64, *SCALE1 = STATS+128, *SHIFT1 = STATS+192;
  float* SUM2 = STATS+256, *SQ2 = STATS+320, *SCALE2 = STATS+384, *SHIFT2 = STATS+448;
  half_t* H1  = (half_t*)alloc((size_t)E * 64 * 2);
  half_t* H2  = (half_t*)alloc((size_t)E * 64 * 2);
  float* HGAT = (float*)alloc((size_t)N * 256 * 4);
  float* GOUT = (float*)alloc((size_t)N * 256 * 4);
  float* G1   = (float*)alloc((size_t)N * 256 * 4);
  float* G2   = (float*)alloc((size_t)N * 64 * 4);
  float* EW   = (float*)alloc((size_t)E * 4 * 4);
  float* ASRC = (float*)alloc((size_t)N * 4 * 4);
  float* ADST = (float*)alloc((size_t)N * 4 * 4);
  float* ESELF= (float*)alloc((size_t)N * 4 * 4);
  float* DENOM= (float*)alloc((size_t)N * 4 * 4);
  unsigned* EMAXK = (unsigned*)alloc((size_t)N * 4 * 4);
  (void)ws_size; (void)n_in; (void)out_size;

  auto pack = [&](const float* s, half_t* d, int K, int NOUT) {
    pack_b_kernel<<<(K * NOUT + 255) / 256, 256, 0, stream>>>(s, d, K, NOUT);
  };
  pack(scW, scWh, 32, 64);
  pack(ec1.W1, e1W1h, 64, 64);  pack(ec1.W2, e1W2h, 64, 64); pack(ec1.W3, e1W3h, 64, 64);
  pack(ec2.W1, e2W1h, 128, 64); pack(ec2.W2, e2W2h, 64, 64); pack(ec2.W3, e2W3h, 64, 64);
  pack(ec3.W1, e3W1h, 128, 64); pack(ec3.W2, e3W2h, 64, 64); pack(ec3.W3, e3W3h, 64, 64);
  pack(g1p.W, g1Wh, 64, 256);
  pack(g2p.W, g2Wh, 256, 256);
  pack(clfW, clfWh, 64, 16);

  const int nodeTiles  = N >> 4;
  const int nodeBlocks = (nodeTiles + 3) / 4;           // 4 waves/block
  const int TPW = 8;
  const int edgeWaves  = ((E >> 4) + TPW - 1) / TPW;
  const int edgeBlocks = (edgeWaves + 3) / 4;

  // shortcut: x_in = x @ Wsc + bsc
  node_gemm_kernel<32, 64, true><<<nodeBlocks, 128, 0, stream>>>(x, scWh, scb, X_IN, N);

  // in-degree for 'mean' aggregation
  hipMemsetAsync(DEG, 0, (size_t)N * 4, stream);
  deg_kernel<<<(E + 255) / 256, 256, 0, stream>>>(dst, DEG, E);

  auto run_edgeconv = [&](const MLP& p, const float* Xin, int DIN,
                          const half_t* W1h, const half_t* W2h, const half_t* W3h,
                          bool mean, const float* res, float* outBuf) {
    hipMemsetAsync(STATS, 0, 512 * 4, stream);
    if (DIN == 32)
      ec_lin1_kernel<32><<<edgeBlocks, 128, 0, stream>>>(Xin, src, dst, W1h, p.b1, H1, SUM1, SQ1, E, TPW);
    else
      ec_lin1_kernel<64><<<edgeBlocks, 128, 0, stream>>>(Xin, src, dst, W1h, p.b1, H1, SUM1, SQ1, E, TPW);
    bn_finalize_kernel<<<1, 64, 0, stream>>>(SUM1, SQ1, p.g1, p.be1, SCALE1, SHIFT1, 1.f / (float)E);
    ec_lin23_kernel<false><<<edgeBlocks, 128, 0, stream>>>(H1, SCALE1, SHIFT1, W2h, p.b2,
                                                           H2, SUM2, SQ2, nullptr, nullptr, E, TPW);
    bn_finalize_kernel<<<1, 64, 0, stream>>>(SUM2, SQ2, p.g2, p.be2, SCALE2, SHIFT2, 1.f / (float)E);
    hipMemsetAsync(AGG, 0, (size_t)N * 64 * 4, stream);
    ec_lin23_kernel<true><<<edgeBlocks, 128, 0, stream>>>(H2, SCALE2, SHIFT2, W3h, p.b3,
                                                          nullptr, nullptr, nullptr, dst, AGG, E, TPW);
    agg_fin_kernel<<<((N * 16) + 255) / 256, 256, 0, stream>>>(AGG, mean ? DEG : nullptr, res, outBuf, N);
  };

  run_edgeconv(ec1, x,    32, e1W1h, e1W2h, e1W3h, true,  nullptr, CUR1);  // x1 = relu(ec1 mean)
  run_edgeconv(ec2, CUR1, 64, e2W1h, e2W2h, e2W3h, false, X_IN,    CUR2);  // out = relu(ec2 sum) + x_in
  run_edgeconv(ec3, CUR2, 64, e3W1h, e3W2h, e3W3h, true,  CUR2,    CUR3);  // out = relu(ec3 mean) + out

  auto run_gat = [&](const float* Xin, int DIN, const half_t* Wh,
                     const GATP& p, bool concat, float* gOut) {
    if (DIN == 64)
      node_gemm_kernel<64, 256, false><<<nodeBlocks, 128, 0, stream>>>(Xin, Wh, nullptr, HGAT, N);
    else
      node_gemm_kernel<256, 256, false><<<nodeBlocks, 128, 0, stream>>>(Xin, Wh, nullptr, HGAT, N);
    gat_att_kernel<<<((N * 4) + 255) / 256, 256, 0, stream>>>(HGAT, p.as, p.ad, ASRC, ADST, ESELF, EMAXK, N);
    gat_edge_max_kernel<<<((E * 4) + 255) / 256, 256, 0, stream>>>(src, dst, ASRC, ADST, EW, EMAXK, E);
    gat_selfdenom_kernel<<<((N * 4) + 255) / 256, 256, 0, stream>>>(ESELF, EMAXK, DENOM, N);
    gat_edge_exp_kernel<<<((E * 4) + 255) / 256, 256, 0, stream>>>(EW, EMAXK, dst, DENOM, E);
    gat_selfout_kernel<<<N, 64, 0, stream>>>(HGAT, ESELF, DENOM, GOUT, N);
    gat_edge_aggr_kernel<<<E, 256, 0, stream>>>(HGAT, EW, DENOM, src, dst, GOUT, E);
    if (concat)
      gat_final_concat_kernel<<<N, 64, 0, stream>>>(GOUT, p.b, gOut, N);
    else
      gat_final_mean_kernel<<<((N * 16) + 255) / 256, 256, 0, stream>>>(GOUT, p.b, gOut, N);
  };

  run_gat(CUR3, 64,  g1Wh, g1p, true,  G1);   // gat1, concat heads -> [N,256], elu
  run_gat(G1,   256, g2Wh, g2p, false, G2);   // gat2, mean heads  -> [N,64], elu

  // classifier -> d_out [N,16]
  node_gemm_kernel<64, 16, true><<<nodeBlocks, 128, 0, stream>>>(G2, clfWh, clfb, (float*)d_out, N);
}
